// BuildingGenerator_29068338659455
// MI455X (gfx1250) — matrix operand order repaired and checked by reading, not compiled, not run on hardware
//
#include <hip/hip_runtime.h>
#include <hip/hip_bf16.h>
#include <stdint.h>

#define NP 5000
#define EP 80000
#define NV 100000
#define EV 600000
#define EC 300000
#define HD 128
#define NC 10

typedef __attribute__((ext_vector_type(16))) __bf16 v16bf;
typedef __attribute__((ext_vector_type(8)))  float  v8f;
typedef __attribute__((ext_vector_type(8)))  unsigned short u16x8;

union Frag { u16x8 h[2]; v16bf b; };

__device__ __forceinline__ unsigned short f2bf(float f) {
  unsigned int u = __float_as_uint(f);
  u += 0x7fffu + ((u >> 16) & 1u);          // round-to-nearest-even (ignoring NaN edge)
  return (unsigned short)(u >> 16);
}
__device__ __forceinline__ float lrelu(float x) { return x > 0.f ? x : 0.01f * x; }

// ---------------------------------------------------------------------------
// Generic tall-skinny GEMM: out(M x 128) = act(A(M x Kp, bf16) @ Wt^T + bias)
// Wt is stored transposed (128 x Kp, bf16). One block = 16 rows, 8 waves each
// owning a 16-column tile; K loop in steps of 32 via v_wmma_f32_16x16x32_bf16.
// MODE 0: out = leaky(.)                 (store)
// MODE 1: out = resid + leaky(.)        (residual store)
// MODE 2: atomicAdd(out[sidx[r]], leaky(.))   (fused segment-sum scatter)
// MODE 3: logits[r] = sum_col tanh(.)*w2[col] + b2   (fused attention head)
// ---------------------------------------------------------------------------
template<int MODE>
__global__ void gemm_wmma(const unsigned short* __restrict__ A,
                          const unsigned short* __restrict__ Wt,
                          const float* __restrict__ bias,
                          float* __restrict__ out,
                          const float* __restrict__ resid,
                          const int* __restrict__ sidx,
                          const float* __restrict__ w2,
                          const float* __restrict__ b2,
                          float* __restrict__ logits,
                          int M, int Kp)
{
  __shared__ float slog[16];
  const int lane = threadIdx.x & 31;
  const int wave = threadIdx.x >> 5;
  const int m    = lane & 15;
  const int kh   = lane >> 4;          // which K-half this lane holds
  const int row0 = blockIdx.x * 16;
  const int n0   = wave * 16;

  if (MODE == 3) {
    if (threadIdx.x < 16) slog[threadIdx.x] = 0.f;
    __syncthreads();
  }

  const int arow = min(row0 + m, M - 1);          // clamp: keep EXEC all-ones
  const unsigned short* ap = A  + (size_t)arow * Kp + kh * 8;
  const unsigned short* bp = Wt + (size_t)(n0 + m) * Kp + kh * 8;

  v8f acc = {};
  for (int kt = 0; kt < Kp; kt += 32) {
    Frag a, b;
    // A 16x32 bf16 fragment: lanes0-15 K[kt..kt+7]+[kt+16..23], lanes16-31 shifted by 8
    a.h[0] = *(const u16x8*)(ap + kt);
    a.h[1] = *(const u16x8*)(ap + kt + 16);
    // B 32x16 bf16 fragment: same K layout as A (lanes0-15 K[kt..kt+7]+[kt+16..23], lanes16-31 shifted by 8)
    b.h[0] = *(const u16x8*)(bp + kt);
    b.h[1] = *(const u16x8*)(bp + kt + 16);
    acc = __builtin_amdgcn_wmma_f32_16x16x32_bf16(false, a.b, false, b.b,
                                                  (short)0, acc, false, false);
  }

  const int   col = n0 + m;
  const float bc  = bias[col];
#pragma unroll
  for (int j = 0; j < 8; ++j) {
    int r = row0 + j + 8 * kh;                 // C layout: VGPR j -> M=j (+8 for hi lanes)
    if (r < M) {
      float val = acc[j] + bc;
      if (MODE == 0)      out[(size_t)r * HD + col] = lrelu(val);
      else if (MODE == 1) out[(size_t)r * HD + col] =
                              resid[(size_t)r * HD + col] + lrelu(val);
      else if (MODE == 2) atomicAdd(&out[(size_t)sidx[r] * HD + col], lrelu(val));
      else                atomicAdd(&slog[j + 8 * kh], tanhf(val) * w2[col]);
    }
  }
  if (MODE == 3) {
    __syncthreads();
    if (threadIdx.x < 16 && row0 + (int)threadIdx.x < M)
      logits[row0 + threadIdx.x] = slog[threadIdx.x] + b2[0];
  }
}

// --- weight prep: transpose (K x 128) fp32 -> (128 x Kp) bf16, zero-padded ---
__global__ void prep_weight(const float* __restrict__ W, unsigned short* __restrict__ Wt,
                            int K, int Kp)
{
  int i = blockIdx.x * 256 + threadIdx.x;
  int total = HD * Kp;
  if (i >= total) return;
  int n = i / Kp, k = i % Kp;
  Wt[(size_t)n * Kp + k] = (k < K) ? f2bf(W[(size_t)k * HD + n]) : (unsigned short)0;
}

// --- input builders (fp32 gather/concat -> bf16 staging rows) ---
__global__ void build_enc(const float* __restrict__ px, const float* __restrict__ pn,
                          unsigned short* __restrict__ A)
{
  int i = blockIdx.x * 256 + threadIdx.x;
  if (i >= NP * 64) return;
  int r = i >> 6, k = i & 63;
  A[i] = f2bf(k < 32 ? px[r * 32 + k] : pn[r * 32 + (k - 32)]);
}

__global__ void build_pairs(const float* __restrict__ X, const int* __restrict__ idxA,
                            const int* __restrict__ idxB, unsigned short* __restrict__ A,
                            int n)
{
  size_t i = (size_t)blockIdx.x * 256 + threadIdx.x;
  if (i >= (size_t)n * 256) return;
  int r = (int)(i >> 8), k = (int)(i & 255);
  int src = (k < 128) ? idxA[r] : idxB[r];
  A[i] = f2bf(X[(size_t)src * HD + (k & 127)]);
}

__global__ void build_upd(const float* __restrict__ x, const float* __restrict__ aggr,
                          const float* __restrict__ cnt, const float* __restrict__ cmean,
                          const float* __restrict__ ccnt, const int* __restrict__ clus,
                          const float* __restrict__ ratio, unsigned short* __restrict__ A)
{
  size_t i = (size_t)blockIdx.x * 256 + threadIdx.x;
  if (i >= (size_t)NP * 384) return;
  int r = (int)(i / 384), k = (int)(i % 384);
  float val;
  if (k < 128)      val = x[(size_t)r * HD + k];
  else if (k < 256) val = aggr[(size_t)r * HD + (k - 128)] / fmaxf(cnt[r], 1.f);
  else {
    int c = clus[r];
    val = (cmean[(size_t)c * HD + (k - 256)] / fmaxf(ccnt[c], 1.f)) * ratio[r];
  }
  A[i] = f2bf(val);
}

__global__ void build_venc(const float* __restrict__ vx, const float* __restrict__ vn,
                           unsigned short* __restrict__ A)
{
  size_t i = (size_t)blockIdx.x * 256 + threadIdx.x;
  if (i >= (size_t)NV * 64) return;
  int r = (int)(i >> 6), k = (int)(i & 63);
  float val = (k < 16) ? vx[(size_t)r * 16 + k]
            : (k < 48) ? vn[(size_t)r * 32 + (k - 16)] : 0.f;
  A[i] = f2bf(val);
}

__global__ void build_att(const float* __restrict__ x, const float* __restrict__ v,
                          const int* __restrict__ cep, const int* __restrict__ cev,
                          unsigned short* __restrict__ A)
{
  size_t i = (size_t)blockIdx.x * 256 + threadIdx.x;
  if (i >= (size_t)EC * 256) return;
  int r = (int)(i >> 8), k = (int)(i & 255);
  float val = (k < 128) ? x[(size_t)cep[r] * HD + k]
                        : v[(size_t)cev[r] * HD + (k - 128)];
  A[i] = f2bf(val);
}

__global__ void build_vm(const float* __restrict__ v, const float* __restrict__ vx,
                         const int* __restrict__ vd, const int* __restrict__ vs,
                         unsigned short* __restrict__ A)
{
  size_t i = (size_t)blockIdx.x * 256 + threadIdx.x;
  if (i >= (size_t)EV * 288) return;
  int r = (int)(i / 288), k = (int)(i % 288);
  int d = vd[r], s = vs[r];
  float val;
  if (k < 128)      val = v[(size_t)d * HD + k];
  else if (k < 256) val = v[(size_t)s * HD + (k - 128)];
  else if (k < 259) val = vx[(size_t)d * 16 + (k - 256)] - vx[(size_t)s * 16 + (k - 256)];
  else              val = 0.f;
  A[i] = f2bf(val);
}

__global__ void build_vupd(const float* __restrict__ v, const float* __restrict__ vaggr,
                           const float* __restrict__ vcnt, unsigned short* __restrict__ A)
{
  size_t i = (size_t)blockIdx.x * 256 + threadIdx.x;
  if (i >= (size_t)NV * 256) return;
  int r = (int)(i >> 8), k = (int)(i & 255);
  float val = (k < 128) ? v[(size_t)r * HD + k]
                        : vaggr[(size_t)r * HD + (k - 128)] / fmaxf(vcnt[r], 1.f);
  A[i] = f2bf(val);
}

// --- small helpers ---
__global__ void count_idx(const int* __restrict__ idx, float* __restrict__ cnt, int n)
{
  int i = blockIdx.x * 256 + threadIdx.x;
  if (i < n) atomicAdd(&cnt[idx[i]], 1.f);
}

__global__ void scatter_rows(const float* __restrict__ X, const int* __restrict__ idx,
                             float* __restrict__ out, int n)
{
  size_t i = (size_t)blockIdx.x * 256 + threadIdx.x;
  if (i >= (size_t)n * HD) return;
  int r = (int)(i >> 7), k = (int)(i & 127);
  atomicAdd(&out[(size_t)idx[r] * HD + k], X[i]);
}

__global__ void softmax_reduce(const float* __restrict__ logits, float* __restrict__ red)
{
  __shared__ float sm[256];
  int tid = threadIdx.x;
  float mx = -3.4e38f;
  for (int i = tid; i < EC; i += 256) mx = fmaxf(mx, logits[i]);
  sm[tid] = mx; __syncthreads();
  for (int s = 128; s > 0; s >>= 1) { if (tid < s) sm[tid] = fmaxf(sm[tid], sm[tid + s]); __syncthreads(); }
  float gmax = sm[0]; __syncthreads();
  float sum = 0.f;
  for (int i = tid; i < EC; i += 256) sum += expf(logits[i] - gmax);
  sm[tid] = sum; __syncthreads();
  for (int s = 128; s > 0; s >>= 1) { if (tid < s) sm[tid] += sm[tid + s]; __syncthreads(); }
  if (tid == 0) { red[0] = gmax; red[1] = sm[0]; }
}

__global__ void attn_apply(const float* __restrict__ logits, const float* __restrict__ red,
                           const float* __restrict__ x, const int* __restrict__ cep,
                           const int* __restrict__ cev, float* __restrict__ v,
                           float* __restrict__ attn_out)
{
  size_t i = (size_t)blockIdx.x * 256 + threadIdx.x;
  if (i >= (size_t)EC * 128) return;
  int e = (int)(i >> 7), k = (int)(i & 127);
  float w = expf(logits[e] - red[0]) / red[1];
  atomicAdd(&v[(size_t)cev[e] * HD + k], w * x[(size_t)cep[e] * HD + k]);
  if (k == 0) attn_out[e] = w;
}

// ---------------------------------------------------------------------------
extern "C" void kernel_launch(void* const* d_in, const int* in_sizes, int n_in,
                              void* d_out, int out_size, void* d_ws, size_t ws_size,
                              hipStream_t stream)
{
  (void)in_sizes; (void)n_in; (void)out_size; (void)ws_size;

  const float* program_x     = (const float*)d_in[0];
  const float* program_noise = (const float*)d_in[1];
  const float* node_ratio    = (const float*)d_in[2];
  const float* voxel_x       = (const float*)d_in[3];
  const float* voxel_noise   = (const float*)d_in[4];
  const int*   pei           = (const int*)d_in[5];
  const int*   node_cluster  = (const int*)d_in[6];
  const int*   vei           = (const int*)d_in[7];
  const int*   cep           = (const int*)d_in[8];
  const int*   cev           = (const int*)d_in[9];
  const float* W_enc      = (const float*)d_in[10];
  const float* b_enc      = (const float*)d_in[11];
  const float* prog_msg_W = (const float*)d_in[12];
  const float* prog_msg_b = (const float*)d_in[13];
  const float* prog_upd_W = (const float*)d_in[14];
  const float* prog_upd_b = (const float*)d_in[15];
  const float* W_venc     = (const float*)d_in[16];
  const float* b_venc     = (const float*)d_in[17];
  const float* vox_msg_W  = (const float*)d_in[18];
  const float* vox_msg_b  = (const float*)d_in[19];
  const float* vox_upd_W  = (const float*)d_in[20];
  const float* vox_upd_b  = (const float*)d_in[21];
  const float* att_W1     = (const float*)d_in[22];
  const float* att_b1     = (const float*)d_in[23];
  const float* att_W2     = (const float*)d_in[24];
  const float* att_b2     = (const float*)d_in[25];

  const int* p_src = pei;       const int* p_dst = pei + EP;
  const int* v_src = vei;       const int* v_dst = vei + EV;

  // workspace carve (peak ~460 MB; big A staging buffer reused across stages)
  char* w = (char*)d_ws;
  auto carve = [&](size_t bytes) -> char* {
    char* p = w; w += (bytes + 255) & ~(size_t)255; return p;
  };
  unsigned short* Abuf    = (unsigned short*)carve((size_t)EV * 288 * 2);
  float* x      = (float*)carve((size_t)NP * HD * 4);
  float* x2     = (float*)carve((size_t)NP * HD * 4);
  float* aggr   = (float*)carve((size_t)NP * HD * 4);
  float* cnt    = (float*)carve((size_t)NP * 4);
  float* cmean  = (float*)carve((size_t)NC * HD * 4);
  float* ccnt   = (float*)carve((size_t)NC * 4);
  float* vbuf   = (float*)carve((size_t)NV * HD * 4);
  float* vaggr  = (float*)carve((size_t)NV * HD * 4);
  float* vcnt   = (float*)carve((size_t)NV * 4);
  float* logits = (float*)carve((size_t)EC * 4);
  float* red    = (float*)carve(256);
  unsigned short* Wt_enc  = (unsigned short*)carve((size_t)HD * 64 * 2);
  unsigned short* Wt_msg  = (unsigned short*)carve((size_t)3 * HD * 256 * 2);
  unsigned short* Wt_upd  = (unsigned short*)carve((size_t)3 * HD * 384 * 2);
  unsigned short* Wt_venc = (unsigned short*)carve((size_t)HD * 64 * 2);
  unsigned short* Wt_att  = (unsigned short*)carve((size_t)HD * 256 * 2);
  unsigned short* Wt_vm   = (unsigned short*)carve((size_t)HD * 288 * 2);
  unsigned short* Wt_vupd = (unsigned short*)carve((size_t)HD * 256 * 2);

  float* out_v   = (float*)d_out;           // NV x 128
  float* out_att = out_v + (size_t)NV * HD; // EC

  auto g1 = [](size_t total) { return dim3((unsigned)((total + 255) / 256)); };

  // --- weight prep (tiny, one-time per call) ---
  prep_weight<<<g1((size_t)HD * 64),  256, 0, stream>>>(W_enc,  Wt_enc,  64,  64);
  for (int s = 0; s < 3; ++s) {
    prep_weight<<<g1((size_t)HD * 256), 256, 0, stream>>>(
        prog_msg_W + (size_t)s * 256 * HD, Wt_msg + (size_t)s * HD * 256, 256, 256);
    prep_weight<<<g1((size_t)HD * 384), 256, 0, stream>>>(
        prog_upd_W + (size_t)s * 384 * HD, Wt_upd + (size_t)s * HD * 384, 384, 384);
  }
  prep_weight<<<g1((size_t)HD * 64),  256, 0, stream>>>(W_venc,    Wt_venc, 48,  64);
  prep_weight<<<g1((size_t)HD * 256), 256, 0, stream>>>(att_W1,    Wt_att,  256, 256);
  prep_weight<<<g1((size_t)HD * 288), 256, 0, stream>>>(vox_msg_W, Wt_vm,   259, 288);
  prep_weight<<<g1((size_t)HD * 256), 256, 0, stream>>>(vox_upd_W, Wt_vupd, 256, 256);

  // --- fixed segment counts ---
  hipMemsetAsync(cnt,  0, (size_t)NP * 4, stream);
  hipMemsetAsync(ccnt, 0, (size_t)NC * 4, stream);
  hipMemsetAsync(vcnt, 0, (size_t)NV * 4, stream);
  count_idx<<<g1(EP), 256, 0, stream>>>(p_dst, cnt, EP);
  count_idx<<<g1(NP), 256, 0, stream>>>(node_cluster, ccnt, NP);
  count_idx<<<g1(EV), 256, 0, stream>>>(v_dst, vcnt, EV);

  // --- program encoder ---
  build_enc<<<g1((size_t)NP * 64), 256, 0, stream>>>(program_x, program_noise, Abuf);
  gemm_wmma<0><<<dim3((NP + 15) / 16), 256, 0, stream>>>(
      Abuf, Wt_enc, b_enc, x, nullptr, nullptr, nullptr, nullptr, nullptr, NP, 64);

  // --- 3 program message-passing steps ---
  float* xa = x; float* xb = x2;
  for (int s = 0; s < 3; ++s) {
    build_pairs<<<g1((size_t)EP * 256), 256, 0, stream>>>(xa, p_dst, p_src, Abuf, EP);
    hipMemsetAsync(aggr, 0, (size_t)NP * HD * 4, stream);
    gemm_wmma<2><<<dim3((EP + 15) / 16), 256, 0, stream>>>(         // msg GEMM + fused seg-sum
        Abuf, Wt_msg + (size_t)s * HD * 256, prog_msg_b + (size_t)s * HD,
        aggr, nullptr, p_dst, nullptr, nullptr, nullptr, EP, 256);
    hipMemsetAsync(cmean, 0, (size_t)NC * HD * 4, stream);
    scatter_rows<<<g1((size_t)NP * HD), 256, 0, stream>>>(xa, node_cluster, cmean, NP);
    build_upd<<<g1((size_t)NP * 384), 256, 0, stream>>>(
        xa, aggr, cnt, cmean, ccnt, node_cluster, node_ratio, Abuf);
    gemm_wmma<1><<<dim3((NP + 15) / 16), 256, 0, stream>>>(         // residual update
        Abuf, Wt_upd + (size_t)s * HD * 384, prog_upd_b + (size_t)s * HD,
        xb, xa, nullptr, nullptr, nullptr, nullptr, NP, 384);
    float* t = xa; xa = xb; xb = t;
  }

  // --- voxel encoder ---
  build_venc<<<g1((size_t)NV * 64), 256, 0, stream>>>(voxel_x, voxel_noise, Abuf);
  gemm_wmma<0><<<dim3((NV + 15) / 16), 256, 0, stream>>>(
      Abuf, Wt_venc, b_venc, vbuf, nullptr, nullptr, nullptr, nullptr, nullptr, NV, 64);

  // --- cross attention (fused logit head; h1 never materialized) ---
  build_att<<<g1((size_t)EC * 256), 256, 0, stream>>>(xa, vbuf, cep, cev, Abuf);
  gemm_wmma<3><<<dim3((EC + 15) / 16), 256, 0, stream>>>(
      Abuf, Wt_att, att_b1, nullptr, nullptr, nullptr, att_W2, att_b2, logits, EC, 256);
  softmax_reduce<<<1, 256, 0, stream>>>(logits, red);
  attn_apply<<<g1((size_t)EC * 128), 256, 0, stream>>>(
      logits, red, xa, cep, cev, vbuf, out_att);

  // --- voxel message passing (fused seg-sum: vm never materialized) ---
  build_vm<<<g1((size_t)EV * 288), 256, 0, stream>>>(vbuf, voxel_x, v_dst, v_src, Abuf);
  hipMemsetAsync(vaggr, 0, (size_t)NV * HD * 4, stream);
  gemm_wmma<2><<<dim3((EV + 15) / 16), 256, 0, stream>>>(
      Abuf, Wt_vm, vox_msg_b, vaggr, nullptr, v_dst, nullptr, nullptr, nullptr, EV, 288);
  build_vupd<<<g1((size_t)NV * 256), 256, 0, stream>>>(vbuf, vaggr, vcnt, Abuf);
  gemm_wmma<1><<<dim3((NV + 15) / 16), 256, 0, stream>>>(          // final residual -> d_out
      Abuf, Wt_vupd, vox_upd_b, out_v, vbuf, nullptr, nullptr, nullptr, nullptr, NV, 256);
}
